// Attention_24489903521914
// MI455X (gfx1250) — matrix operand order repaired and checked by reading, not compiled, not run on hardware
//
#include <hip/hip_runtime.h>
#include <hip/hip_bf16.h>

typedef _Float16 f16;
typedef __attribute__((ext_vector_type(16))) _Float16 v16h;
typedef __attribute__((ext_vector_type(8)))  float    v8f;
typedef __attribute__((__vector_size__(16))) int      vi4;   // matches builtin param

#define DIM   1024
#define HEADS 16
#define HD    64
#define ROT   32
#define SEQN  2048
#define BATCH 2
#define BN    (BATCH * SEQN)   // 4096 rows

union AFrag { v16h v; uint4 q[2]; };

#if defined(__has_builtin)
# if __has_builtin(__builtin_amdgcn_global_load_async_to_lds_b128)
#  define HAVE_ASYNC_LDS 1
# endif
# if __has_builtin(__builtin_amdgcn_s_wait_asynccnt)
#  define HAVE_WAIT_ASYNC 1
# endif
#endif

// ---------------------------------------------------------------------------
// f32 -> f16 cast
// ---------------------------------------------------------------------------
__global__ __launch_bounds__(256) void cast_f32_f16(const float* __restrict__ src,
                                                    f16* __restrict__ dst, int n) {
  int i = blockIdx.x * blockDim.x + threadIdx.x;
  if (i < n) dst[i] = (f16)src[i];
}

// ---------------------------------------------------------------------------
// Tiled WMMA GEMM: C[M,N] = A[M,K] @ B[K,N], f16 in, f32 accumulate.
// 256 threads (8 waves), block tile 128x128, wave tile 32x64, K-step 64.
// STORE_MODE 0: f16 output; 1: f32 output.
// ---------------------------------------------------------------------------
#define LDSA 72   // LDS row stride in f16 (144B, 16B aligned, conflict-friendly)

template <int STORE_MODE>
__global__ __launch_bounds__(256)
void gemm_wmma_f16(const f16* __restrict__ A, const f16* __restrict__ B,
                   void* __restrict__ Cout, int M, int N, int K) {
  __shared__ __align__(16) f16 As[128 * LDSA];
  __shared__ __align__(16) f16 Bt[128 * LDSA];   // B transposed: Bt[n][k]

  const int tid  = threadIdx.x;
  const int lane = tid & 31;
  const int wave = tid >> 5;
  const int r    = lane & 15;
  const int half = lane >> 4;
  const int m0 = blockIdx.y * 128;
  const int n0 = blockIdx.x * 128;
  const int mstrip = (wave & 3) * 32;   // wave's M offset in block tile
  const int nstrip = (wave >> 2) * 64;  // wave's N offset in block tile

  v8f acc[2][4] = {};

  const int arow = tid >> 1;            // 0..127
  const int acol = (tid & 1) * 32;      // 0 / 32
  const int brow = tid >> 2;            // 0..63  (k)
  const int bcol = (tid & 3) * 32;      // 0/32/64/96 (n)

  for (int k0 = 0; k0 < K; k0 += 64) {
    // ---- stage A 128x64 tile into LDS (async path if available) ----
    const f16* ag = A + (size_t)(m0 + arow) * K + k0 + acol;
    f16* al = &As[arow * LDSA + acol];
#if defined(HAVE_ASYNC_LDS)
#pragma unroll
    for (int qq = 0; qq < 4; ++qq)
      __builtin_amdgcn_global_load_async_to_lds_b128(
          (__attribute__((address_space(1))) vi4*)(ag + qq * 8),
          (__attribute__((address_space(3))) vi4*)(al + qq * 8), 0, 0);
#else
#pragma unroll
    for (int qq = 0; qq < 4; ++qq)
      *(uint4*)(al + qq * 8) = *(const uint4*)(ag + qq * 8);
#endif
    // ---- stage B 64x128 tile transposed into Bt[n][k] ----
    const f16* bg = B + (size_t)(k0 + brow) * N + n0 + bcol;
    uint4 bq[4];
#pragma unroll
    for (int qq = 0; qq < 4; ++qq) bq[qq] = *(const uint4*)(bg + qq * 8);
    const f16* be = (const f16*)bq;
#pragma unroll
    for (int e = 0; e < 32; ++e) Bt[(bcol + e) * LDSA + brow] = be[e];

#if defined(HAVE_ASYNC_LDS)
# if defined(HAVE_WAIT_ASYNC)
    __builtin_amdgcn_s_wait_asynccnt(0);
# else
    __asm__ volatile("s_wait_asynccnt 0" ::: "memory");
# endif
#endif
    __syncthreads();

    // ---- compute: 2 K-chunks of 32, 2 M-frags x 4 N-frags ----
    const int kb  = half ? 8 : 0;    // unused; A frags also use kbb
    const int kbb = half ? 16 : 0;   // B-frag K sub-offset
#pragma unroll
    for (int kc = 0; kc < 2; ++kc) {
      AFrag af[2];
#pragma unroll
      for (int mi = 0; mi < 2; ++mi) {
        const f16* ap = &As[(mstrip + mi * 16 + r) * LDSA + kc * 32];
        af[mi].q[0] = *(const uint4*)(ap + kbb);
        af[mi].q[1] = *(const uint4*)(ap + kbb + 8);
      }
#pragma unroll
      for (int nb = 0; nb < 4; ++nb) {
        AFrag bf;
        const f16* bp = &Bt[(nstrip + nb * 16 + r) * LDSA + kc * 32 + kbb];
        bf.q[0] = *(const uint4*)(bp);
        bf.q[1] = *(const uint4*)(bp + 8);
#pragma unroll
        for (int mi = 0; mi < 2; ++mi)
          acc[mi][nb] = __builtin_amdgcn_wmma_f32_16x16x32_f16(
              false, af[mi].v, false, bf.v, (short)0, acc[mi][nb], false, false);
      }
    }
    __syncthreads();
  }

  // ---- epilogue (C layout: VGPR g -> row g + 8*half, col = lane%16) ----
#pragma unroll
  for (int mi = 0; mi < 2; ++mi) {
#pragma unroll
    for (int nb = 0; nb < 4; ++nb) {
#pragma unroll
      for (int g = 0; g < 8; ++g) {
        size_t row = m0 + mstrip + mi * 16 + g + half * 8;
        size_t col = n0 + nstrip + nb * 16 + r;
        float v = acc[mi][nb][g];
        if (STORE_MODE == 1) ((float*)Cout)[row * N + col] = v;
        else                 ((f16*)Cout)[row * N + col]   = (f16)v;
      }
    }
  }
}

// ---------------------------------------------------------------------------
// RoPE helpers: cos/sin on the fly (inv_freq[k] = 10000^(-k/16))
// ---------------------------------------------------------------------------
__device__ __forceinline__ float rope_val(float t, float other, int d, int pos) {
  int k = d & 15;
  float invf = __expf(-(float)k * 0.5756462732485114f);  // ln(10000)/16
  float ang = (float)pos * invf;
  return t * __cosf(ang) + other * __sinf(ang);
}

__global__ __launch_bounds__(256)
void rope_scatter_q(const f16* __restrict__ Qtmp, f16* __restrict__ Qb) {
  size_t idx = (size_t)blockIdx.x * blockDim.x + threadIdx.x;  // B*SEQ*1024
  int d = idx & 63;
  int h = (int)((idx >> 6) & 15);
  size_t bn = idx >> 10;
  int pos = (int)(bn & (SEQN - 1));
  int b   = (int)(bn >> 11);
  float t = (float)Qtmp[bn * 1024 + h * 64 + d];
  float val;
  if (d < ROT) {
    float other = (d < 16) ? -(float)Qtmp[bn * 1024 + h * 64 + d + 16]
                           :  (float)Qtmp[bn * 1024 + h * 64 + d - 16];
    val = rope_val(t, other, d, pos);
  } else {
    val = t;
  }
  val *= 0.125f;  // 1/sqrt(HEAD_DIM)
  Qb[(((size_t)(b * HEADS + h)) * SEQN + pos) * HD + d] = (f16)val;
}

__global__ __launch_bounds__(256)
void rope_scatter_kv(const f16* __restrict__ KVtmp,
                     f16* __restrict__ Kb, f16* __restrict__ Vt) {
  size_t idx = (size_t)blockIdx.x * blockDim.x + threadIdx.x;  // B*SEQ*2048
  int col = (int)(idx & 2047);
  size_t bn = idx >> 11;
  int pos = (int)(bn & (SEQN - 1));
  int b   = (int)(bn >> 11);
  float t = (float)KVtmp[bn * 2048 + col];
  if (col < 1024) {
    int d = col & 63, h = col >> 6;
    float val;
    if (d < ROT) {
      float other = (d < 16) ? -(float)KVtmp[bn * 2048 + h * 64 + d + 16]
                             :  (float)KVtmp[bn * 2048 + h * 64 + d - 16];
      val = rope_val(t, other, d, pos);
    } else {
      val = t;
    }
    Kb[(((size_t)(b * HEADS + h)) * SEQN + pos) * HD + d] = (f16)val;
  } else {
    int c = col - 1024;
    int d = c & 63, h = c >> 6;
    Vt[(((size_t)(b * HEADS + h)) * HD + d) * SEQN + pos] = (f16)t;
  }
}

// ---------------------------------------------------------------------------
// Flash attention (transposed form): one wave per (b,h, 16-query tile).
// S^T = K @ Q^T  (rows = keys j, cols = queries i)  -> softmax is per-lane
// O^T = V^T @ P^T (rows = head dim d, cols = queries i)
// KV block = 32 keys (two 16-row S^T tiles -> one 32-K P^T fragment).
// ---------------------------------------------------------------------------
#define LDSS 40

__global__ __launch_bounds__(128)
void flash_attn(const f16* __restrict__ Qb, const f16* __restrict__ Kb,
                const f16* __restrict__ Vt, f16* __restrict__ Ob) {
  __shared__ __align__(16) f16 Plds[4 * 16 * LDSS];  // per-wave P scratch [i][j]

  const int lane = threadIdx.x & 31;
  const int wave = threadIdx.x >> 5;
  const int r    = lane & 15;
  const int half = lane >> 4;
  const int kb   = half ? 8 : 0;     // unused; A frags also use kbb
  const int kbb  = half ? 16 : 0;    // B-frag K sub-offset

  const int TILES = SEQN / 16;                 // 128
  const int wg   = blockIdx.x * 4 + wave;
  const int tile = wg % TILES;
  const int bh   = wg / TILES;                 // 0..31
  const int h    = bh % HEADS;
  const int b    = bh / HEADS;
  const int i0   = __builtin_amdgcn_readfirstlane(tile * 16);  // wave-uniform
  const int imax = i0 + 15;

  const f16* Q = Qb + (size_t)bh * SEQN * HD;
  const f16* K = Kb + (size_t)bh * SEQN * HD;
  const f16* V = Vt + (size_t)bh * HD * SEQN;

  // Q as two B-fragments of Q^T (cols = queries, K-dim = head dim chunks)
  AFrag qf[2];
  {
    const f16* qrow = Q + (size_t)(i0 + r) * HD;
    qf[0].q[0] = *(const uint4*)(qrow + kbb);
    qf[0].q[1] = *(const uint4*)(qrow + kbb + 8);
    qf[1].q[0] = *(const uint4*)(qrow + 32 + kbb);
    qf[1].q[1] = *(const uint4*)(qrow + 32 + kbb + 8);
  }

  float mcur = -3.0e38f;   // per-lane: query i = i0 + r
  float lcur = 0.0f;
  v8f o[4] = {};           // O^T chunks: rows d = cc*16 + g + 8*half, col i = r

  f16* pl = &Plds[wave * 16 * LDSS];

  for (int j0 = 0; j0 <= imax; j0 += 32) {     // uniform trip count
    v8f s[2] = {};
#pragma unroll
    for (int t = 0; t < 2; ++t) {
      const int jt = j0 + t * 16;
      if (jt <= imax) {                        // SGPR-uniform condition
#pragma unroll
        for (int c = 0; c < 2; ++c) {          // head-dim chunks
          AFrag kf;                            // A frag: K rows j, K-dim = d
          const f16* krow = K + (size_t)(jt + r) * HD + c * 32;
          kf.q[0] = *(const uint4*)(krow + kbb);
          kf.q[1] = *(const uint4*)(krow + kbb + 8);
          s[t] = __builtin_amdgcn_wmma_f32_16x16x32_f16(
              false, kf.v, false, qf[c].v, (short)0, s[t], false, false);
        }
#pragma unroll
        for (int g = 0; g < 8; ++g) {          // causal: j > i -> -inf
          int j = jt + g + half * 8;
          int i = i0 + r;
          s[t][g] = (j > i) ? -3.0e38f : s[t][g];
        }
      } else {
#pragma unroll
        for (int g = 0; g < 8; ++g) s[t][g] = -3.0e38f;
      }
    }

    // per-lane online softmax (lane = one query column)
    float mx = s[0][0];
#pragma unroll
    for (int g = 1; g < 8; ++g) mx = fmaxf(mx, s[0][g]);
#pragma unroll
    for (int g = 0; g < 8; ++g) mx = fmaxf(mx, s[1][g]);
    mx = fmaxf(mx, __shfl_xor(mx, 16, 32));    // merge j halves
    float mnew = fmaxf(mcur, mx);
    float corr = __expf(mcur - mnew);
    float rs = 0.0f;
#pragma unroll
    for (int t = 0; t < 2; ++t)
#pragma unroll
      for (int g = 0; g < 8; ++g) {
        float p = __expf(s[t][g] - mnew);
        rs += p;
        // store P[i][j] (B-frag source layout for P^T)
        pl[r * LDSS + t * 16 + g + half * 8] = (f16)p;
      }
    rs += __shfl_xor(rs, 16, 32);
    lcur = lcur * corr + rs;
    mcur = mnew;
#pragma unroll
    for (int cc = 0; cc < 4; ++cc)
#pragma unroll
      for (int g = 0; g < 8; ++g) o[cc][g] *= corr;

    // DS is in-order within a wave; wait + compiler barrier before re-read
    __asm__ volatile("s_wait_dscnt 0" ::: "memory");

    AFrag pf;                                  // P^T as B fragment (32j x 16i)
    pf.q[0] = *(const uint4*)(&pl[r * LDSS + kbb]);
    pf.q[1] = *(const uint4*)(&pl[r * LDSS + kbb + 8]);

    // O^T += V^T @ P^T  (Vt rows d contiguous over j -> A fragments)
#pragma unroll
    for (int cc = 0; cc < 4; ++cc) {
      AFrag vf;
      const f16* vrow = V + (size_t)(cc * 16 + r) * SEQN + j0;
      vf.q[0] = *(const uint4*)(vrow + kbb);
      vf.q[1] = *(const uint4*)(vrow + kbb + 8);
      o[cc] = __builtin_amdgcn_wmma_f32_16x16x32_f16(
          false, vf.v, false, pf.v, (short)0, o[cc], false, false);
    }
  }

  // epilogue: lane holds query i = i0 + r, 8 consecutive d per (cc,half)
  const float inv = 1.0f / lcur;
  const int i = i0 + r;
#pragma unroll
  for (int cc = 0; cc < 4; ++cc) {
    union { uint4 q; f16 e[8]; } pack;
#pragma unroll
    for (int g = 0; g < 8; ++g) pack.e[g] = (f16)(o[cc][g] * inv);
    f16* dst = Ob + ((size_t)(b * SEQN + i)) * (HEADS * HD)
                  + h * HD + cc * 16 + half * 8;
    *(uint4*)dst = pack.q;
  }
}

// ---------------------------------------------------------------------------
// Launcher
// ---------------------------------------------------------------------------
extern "C" void kernel_launch(void* const* d_in, const int* in_sizes, int n_in,
                              void* d_out, int out_size, void* d_ws, size_t ws_size,
                              hipStream_t stream) {
  const float* x   = (const float*)d_in[0];  // [2,2048,1024]
  const float* wq  = (const float*)d_in[1];  // [1024,1024]
  const float* wkv = (const float*)d_in[2];  // [1024,2048]
  const float* wo  = (const float*)d_in[3];  // [1024,1024]
  float* out = (float*)d_out;                // [2,2048,1024] f32

  const size_t MB = 1u << 20;
  char* ws = (char*)d_ws;
  f16* xh    = (f16*)(ws + 0 * MB);    // 8MB: x f16 (4096x1024)
  f16* wqh   = (f16*)(ws + 8 * MB);    // 2MB
  f16* wkvh  = (f16*)(ws + 10 * MB);   // 4MB
  f16* woh   = (f16*)(ws + 14 * MB);   // 2MB
  f16* Qbuf  = (f16*)(ws + 16 * MB);   // 8MB: [b,h,seq,d]
  f16* Kbuf  = (f16*)(ws + 24 * MB);   // 8MB: [b,h,seq,d]
  f16* Vtb   = (f16*)(ws + 32 * MB);   // 8MB: [b,h,d,seq]
  f16* Obuf  = (f16*)(ws + 40 * MB);   // 8MB: [bn, h*64+d]
  f16* Qtmp  = (f16*)(ws + 48 * MB);   // 8MB: 4096x1024
  f16* KVtmp = (f16*)(ws + 56 * MB);   // 16MB: 4096x2048

  // 1) casts
  cast_f32_f16<<<(BN * DIM) / 256, 256, 0, stream>>>(x, xh, BN * DIM);
  cast_f32_f16<<<(DIM * DIM) / 256, 256, 0, stream>>>(wq, wqh, DIM * DIM);
  cast_f32_f16<<<(DIM * 2 * DIM) / 256, 256, 0, stream>>>(wkv, wkvh, DIM * 2 * DIM);
  cast_f32_f16<<<(DIM * DIM) / 256, 256, 0, stream>>>(wo, woh, DIM * DIM);

  // 2) projections (block tile 128x128)
  gemm_wmma_f16<0><<<dim3(DIM / 128, BN / 128), 256, 0, stream>>>(
      xh, wqh, (void*)Qtmp, BN, DIM, DIM);
  gemm_wmma_f16<0><<<dim3((2 * DIM) / 128, BN / 128), 256, 0, stream>>>(
      xh, wkvh, (void*)KVtmp, BN, 2 * DIM, DIM);

  // 3) RoPE + scatter (V transposed)
  rope_scatter_q<<<(BN * DIM) / 256, 256, 0, stream>>>(Qtmp, Qbuf);
  rope_scatter_kv<<<(BN * 2 * DIM) / 256, 256, 0, stream>>>(KVtmp, Kbuf, Vtb);

  // 4) causal flash attention: 2*16*128 wave-tiles / 4 waves per block
  flash_attn<<<(BATCH * HEADS * (SEQN / 16)) / 4, 128, 0, stream>>>(
      Qbuf, Kbuf, Vtb, Obuf);

  // 5) output projection (f32 out)
  gemm_wmma_f16<1><<<dim3(DIM / 128, BN / 128), 256, 0, stream>>>(
      Obuf, woh, (void*)out, BN, DIM, DIM);
}
